// MyLinear_40321152975111
// MI455X (gfx1250) — matrix operand order repaired and checked by reading, not compiled, not run on hardware
//
#include <hip/hip_runtime.h>

typedef __bf16 bf16;
typedef __attribute__((ext_vector_type(16))) __bf16 v16bf;
typedef __attribute__((ext_vector_type(8)))  __bf16 v8bf;
typedef __attribute__((ext_vector_type(4)))  __bf16 v4bf;
typedef __attribute__((ext_vector_type(8)))  float  v8f;
typedef __attribute__((ext_vector_type(4)))  float  f32x4;

#define MDIM 4096
#define NDIM 4096
#define KDIM 4096

constexpr int TM  = 128;   // block tile M
constexpr int TN  = 128;   // block tile N
constexpr int TK  = 32;    // k-step (matches bf16 WMMA K)
constexpr int AST = 40;    // padded LDS row stride in bf16 elements (80B, 16B-aligned)
constexpr int NK  = KDIM / TK;

union V16 { v16bf v; v8bf h[2]; };

__global__ __launch_bounds__(256) void linear_bf16x3_wmma(
    const float* __restrict__ X,   // [4096, 4096] row-major (b, k)
    const float* __restrict__ W,   // [4096, 4096] row-major (k, n)
    const float* __restrict__ Bv,  // [4096] bias
    float* __restrict__ O)         // [4096, 4096] row-major (b, n)
{
  // hi/lo bf16 planes.  A: [row][k] (row-major).  B: transposed [n][k].
  __shared__ bf16 Ah[TM * AST];
  __shared__ bf16 Al[TM * AST];
  __shared__ bf16 Bh[TN * AST];
  __shared__ bf16 Bl[TN * AST];

  const int tid  = threadIdx.x;
  const int lane = tid & 31;
  const int wave = tid >> 5;
  const int wm   = wave & 1;     // wave row: 2 x 64 rows
  const int wn   = wave >> 1;    // wave col: 4 x 32 cols
  const int half = lane >> 4;    // lane group 0/1
  const int lr   = lane & 15;

  const int bm = blockIdx.y * TM;
  const int bn = blockIdx.x * TN;

  f32x4 ar[4], br[4];            // global prefetch registers

  auto loadA = [&](int kk) {
#pragma unroll
    for (int j = 0; j < 4; ++j) {
      int idx = tid + j * 256;           // 1024 float4s = 128 rows x 8
      int row = idx >> 3, c4 = idx & 7;
      ar[j] = *(const f32x4*)(X + (size_t)(bm + row) * KDIM + kk + c4 * 4);
    }
  };
  auto loadB = [&](int kk) {
#pragma unroll
    for (int j = 0; j < 4; ++j) {
      int idx = tid + j * 256;           // 1024 float4s = 32 rows x 32
      int row = idx >> 5, c4 = idx & 31;
      br[j] = *(const f32x4*)(W + (size_t)(kk + row) * NDIM + bn + c4 * 4);
    }
  };
  auto storeA = [&]() {
#pragma unroll
    for (int j = 0; j < 4; ++j) {
      int idx = tid + j * 256;
      int row = idx >> 3, c4 = idx & 7;
      v4bf h, l;
#pragma unroll
      for (int e = 0; e < 4; ++e) {
        float f  = ar[j][e];
        bf16 hh  = (bf16)f;
        h[e] = hh;
        l[e] = (bf16)(f - (float)hh);
      }
      *(v4bf*)(Ah + row * AST + c4 * 4) = h;
      *(v4bf*)(Al + row * AST + c4 * 4) = l;
    }
  };
  auto storeB = [&]() {                  // transpose into [n][k] planes
#pragma unroll
    for (int j = 0; j < 4; ++j) {
      int idx  = tid + j * 256;
      int krow = idx >> 5, c4 = idx & 31;
#pragma unroll
      for (int e = 0; e < 4; ++e) {
        int n   = c4 * 4 + e;
        float f = br[j][e];
        bf16 hh = (bf16)f;
        Bh[n * AST + krow] = hh;
        Bl[n * AST + krow] = (bf16)(f - (float)hh);
      }
    }
  };

  v8f acc[4][2];
#pragma unroll
  for (int i = 0; i < 4; ++i)
#pragma unroll
    for (int j = 0; j < 2; ++j)
#pragma unroll
      for (int e = 0; e < 8; ++e) acc[i][j][e] = 0.0f;

  auto compute = [&]() {
    V16 bh[2], bl[2];
#pragma unroll
    for (int j = 0; j < 2; ++j) {
      // B fragment: column = wn*32 + j*16 + lr ; elems 0..7 <- K=half*8.., elems 8..15 <- K=16+half*8..
      const bf16* ph = Bh + (wn * 32 + j * 16 + lr) * AST + half * 8;
      const bf16* pl = Bl + (wn * 32 + j * 16 + lr) * AST + half * 8;
      bh[j].h[0] = *(const v8bf*)(ph);
      bh[j].h[1] = *(const v8bf*)(ph + 16);
      bl[j].h[0] = *(const v8bf*)(pl);
      bl[j].h[1] = *(const v8bf*)(pl + 16);
    }
#pragma unroll
    for (int i = 0; i < 4; ++i) {
      // A fragment: row = wm*64 + i*16 + lr ; elems 0..7 <- K=half*8.., elems 8..15 <- K=16+half*8..
      const bf16* pa = Ah + (wm * 64 + i * 16 + lr) * AST + half * 8;
      const bf16* pb = Al + (wm * 64 + i * 16 + lr) * AST + half * 8;
      V16 ah, al;
      ah.h[0] = *(const v8bf*)(pa);
      ah.h[1] = *(const v8bf*)(pa + 16);
      al.h[0] = *(const v8bf*)(pb);
      al.h[1] = *(const v8bf*)(pb + 16);
#pragma unroll
      for (int j = 0; j < 2; ++j) {
        acc[i][j] = __builtin_amdgcn_wmma_f32_16x16x32_bf16(
            false, ah.v, false, bh[j].v, (short)0, acc[i][j], false, false);
        acc[i][j] = __builtin_amdgcn_wmma_f32_16x16x32_bf16(
            false, ah.v, false, bl[j].v, (short)0, acc[i][j], false, false);
        acc[i][j] = __builtin_amdgcn_wmma_f32_16x16x32_bf16(
            false, al.v, false, bh[j].v, (short)0, acc[i][j], false, false);
      }
    }
  };

  // ---- pipeline: prefetch next tile into regs while computing on LDS ----
  loadA(0);
  loadB(0);
  storeA();
  storeB();
  __syncthreads();

  for (int kt = 1; kt <= NK; ++kt) {
    if (kt < NK) { loadA(kt * TK); loadB(kt * TK); }
    compute();
    __syncthreads();
    if (kt < NK) {
      storeA();
      storeB();
      __syncthreads();
    }
  }

  // ---- epilogue: bias add + store (C/D layout: lane<16 -> M=v, lane>=16 -> M=v+8) ----
#pragma unroll
  for (int j = 0; j < 2; ++j) {
    int n = bn + wn * 32 + j * 16 + lr;
    float bv = Bv[n];
#pragma unroll
    for (int i = 0; i < 4; ++i) {
      int rbase = bm + wm * 64 + i * 16 + half * 8;
#pragma unroll
      for (int v = 0; v < 8; ++v) {
        O[(size_t)(rbase + v) * NDIM + n] = acc[i][j][v] + bv;
      }
    }
  }
}

extern "C" void kernel_launch(void* const* d_in, const int* in_sizes, int n_in,
                              void* d_out, int out_size, void* d_ws, size_t ws_size,
                              hipStream_t stream) {
  const float* x  = (const float*)d_in[0];   // [4096,4096]
  const float* w  = (const float*)d_in[1];   // [1,4096,4096]
  const float* bv = (const float*)d_in[2];   // [1,4096]
  float* out      = (float*)d_out;

  dim3 grid(NDIM / TN, MDIM / TM);           // 32 x 32 workgroups
  linear_bf16x3_wmma<<<grid, dim3(256), 0, stream>>>(x, w, bv, out);
}
